// TPDecoder_56556129353848
// MI455X (gfx1250) — matrix operand order repaired and checked by reading, not compiled, not run on hardware
//
#include <hip/hip_runtime.h>
#include <hip/hip_bf16.h>

// ---------------------------------------------------------------------------
// Problem constants (match reference)
// ---------------------------------------------------------------------------
constexpr int NB  = 16;          // batch
constexpr int NS  = 200;         // seq len
constexpr int ND  = 1024;        // model dim
constexpr int NL  = 6;           // layers
constexpr int NH  = 16;          // heads
constexpr int NHD = 64;          // head dim
constexpr int NV  = 32000;       // vocab
constexpr int NF  = 4096;        // ffn hidden
constexpr int NM  = NB * NS;     // 3200 rows

typedef __attribute__((ext_vector_type(16))) __bf16 v16bf;
typedef __attribute__((ext_vector_type(8)))  __bf16 v8bf;
typedef __attribute__((ext_vector_type(8)))  float  v8f;

// ---------------------------------------------------------------------------
// bf16 WMMA GEMM:  C[M,N] = A[M,K] * W[K,N] + bias  (RELU as template param)
// block tile 128x64, K-step 32, 8 waves (4 along M x 2 along N), wave 32x32
// LDS double-buffered: global f32 loads for stage k+1 overlap WMMAs of stage k
// Requires: M%128==0, N%64==0, K%32==0 (true for every call below)
// ---------------------------------------------------------------------------
#define LDK 40   // padded K stride (bf16 elements); 40*2=80B, rows 16B-aligned

template<int RELU>
__global__ __launch_bounds__(256) void gemm_bf16_wmma(
    const float* __restrict__ A, const float* __restrict__ W,
    const float* __restrict__ bias, float* __restrict__ C,
    int K, int N)
{
    __shared__ __bf16 As[2][128 * LDK];
    __shared__ __bf16 Bs[2][64 * LDK];   // stored [n][k]

    const int tid  = threadIdx.x;
    const int lane = tid & 31;
    const int wave = tid >> 5;
    const int l16  = lane & 15;
    const int half = lane >> 4;

    const int m0 = (wave & 3) * 32;   // wave M offset in block
    const int n0 = (wave >> 2) * 32;  // wave N offset in block
    const int blockM = blockIdx.y * 128;
    const int blockN = blockIdx.x * 64;

    v8f acc[2][2];
#pragma unroll
    for (int mi = 0; mi < 2; ++mi)
#pragma unroll
        for (int ni = 0; ni < 2; ++ni)
#pragma unroll
            for (int j = 0; j < 8; ++j) acc[mi][ni][j] = 0.0f;

    // staging registers for one (A,B) tile
    float4 ra[4], rb[2];

    auto loadGlobal = [&](int k0) {
#pragma unroll
        for (int it = 0; it < 4; ++it) {
            int i  = it * 1024 + tid * 4;
            int r  = i >> 5;
            int kk = i & 31;
            ra[it] = *(const float4*)&A[(size_t)(blockM + r) * K + k0 + kk];
        }
#pragma unroll
        for (int it = 0; it < 2; ++it) {
            int i  = it * 1024 + tid * 4;
            int kk = i >> 6;
            int n  = i & 63;
            rb[it] = *(const float4*)&W[(size_t)(k0 + kk) * N + blockN + n];
        }
    };
    auto storeLds = [&](int buf) {
#pragma unroll
        for (int it = 0; it < 4; ++it) {
            int i  = it * 1024 + tid * 4;
            int r  = i >> 5;
            int kk = i & 31;
            __bf16* dst = &As[buf][r * LDK + kk];
            dst[0] = (__bf16)ra[it].x; dst[1] = (__bf16)ra[it].y;
            dst[2] = (__bf16)ra[it].z; dst[3] = (__bf16)ra[it].w;
        }
#pragma unroll
        for (int it = 0; it < 2; ++it) {
            int i  = it * 1024 + tid * 4;
            int kk = i >> 6;
            int n  = i & 63;
            Bs[buf][(n + 0) * LDK + kk] = (__bf16)rb[it].x;
            Bs[buf][(n + 1) * LDK + kk] = (__bf16)rb[it].y;
            Bs[buf][(n + 2) * LDK + kk] = (__bf16)rb[it].z;
            Bs[buf][(n + 3) * LDK + kk] = (__bf16)rb[it].w;
        }
    };

    // prologue: stage tile 0
    loadGlobal(0);
    storeLds(0);
    __syncthreads();

    int buf = 0;
    for (int k0 = 0; k0 < K; k0 += 32) {
        const bool hasNext = (k0 + 32) < K;
        if (hasNext) loadGlobal(k0 + 32);   // overlap with WMMAs below

        // ---- fragment loads (ISA 16-bit wave32 layouts) ----
        // A 16x32: lanes 0-15 hold K{0..7,16..23}, lanes 16-31 K{8..15,24..31}
        v16bf afrag[2], bfrag[2];
#pragma unroll
        for (int mi = 0; mi < 2; ++mi) {
            const __bf16* base = &As[buf][(m0 + mi * 16 + l16) * LDK + half * 8];
            v8bf lo = *(const v8bf*)(base);
            v8bf hi = *(const v8bf*)(base + 16);
#pragma unroll
            for (int e = 0; e < 8; ++e) { afrag[mi][e] = lo[e]; afrag[mi][e + 8] = hi[e]; }
        }
        // B 32x16: lanes 0-15 hold K{0..7,16..23} of col N=lane, lanes 16-31 K{8..15,24..31}
#pragma unroll
        for (int ni = 0; ni < 2; ++ni) {
            const __bf16* base = &Bs[buf][(n0 + ni * 16 + l16) * LDK + half * 8];
            v8bf lo = *(const v8bf*)(base);
            v8bf hi = *(const v8bf*)(base + 16);
#pragma unroll
            for (int e = 0; e < 8; ++e) { bfrag[ni][e] = lo[e]; bfrag[ni][e + 8] = hi[e]; }
        }

#pragma unroll
        for (int mi = 0; mi < 2; ++mi)
#pragma unroll
            for (int ni = 0; ni < 2; ++ni)
                acc[mi][ni] = __builtin_amdgcn_wmma_f32_16x16x32_bf16(
                    false, afrag[mi], false, bfrag[ni],
                    (short)0, acc[mi][ni], false, false);

        if (hasNext) storeLds(buf ^ 1);     // write other buffer; no race w/ reads
        __syncthreads();
        buf ^= 1;
    }

    // ---- epilogue: bias (+ compile-time ReLU), f32 store ----
    // D 16x16 f32: VGPR j -> M=j (lanes 0-15) / M=8+j (lanes 16-31), N=lane%16
#pragma unroll
    for (int mi = 0; mi < 2; ++mi) {
#pragma unroll
        for (int ni = 0; ni < 2; ++ni) {
            int col = blockN + n0 + ni * 16 + l16;
            float bv = bias[col];
#pragma unroll
            for (int j = 0; j < 8; ++j) {
                int row = blockM + m0 + mi * 16 + half * 8 + j;
                float v = acc[mi][ni][j] + bv;
                if (RELU) v = v > 0.0f ? v : 0.0f;
                C[(size_t)row * N + col] = v;
            }
        }
    }
}

// ---------------------------------------------------------------------------
// Embedding gather:  x0 = tok_emb[id]*sqrt(D) + pe[s]
// ---------------------------------------------------------------------------
__global__ __launch_bounds__(256) void embed_kernel(
    const int* __restrict__ ids, const float* __restrict__ te,
    const float* __restrict__ pe, float* __restrict__ x0)
{
    int row = blockIdx.x;            // 0..NM-1
    int s   = row % NS;
    int id  = ids[row];
    int d   = threadIdx.x * 4;       // 256*4 == 1024 == ND
    const float4 t = *(const float4*)&te[(size_t)id * ND + d];
    const float4 p = *(const float4*)&pe[(size_t)s * ND + d];
    float4 o;
    o.x = t.x * 32.0f + p.x;  o.y = t.y * 32.0f + p.y;
    o.z = t.z * 32.0f + p.z;  o.w = t.w * 32.0f + p.w;
    *(float4*)&x0[(size_t)row * ND + d] = o;
}

// x = x0 * (r + 1)
__global__ __launch_bounds__(256) void rgate_kernel(
    const float* __restrict__ x0, const float* __restrict__ r, float* __restrict__ x)
{
    size_t i = (size_t)blockIdx.x * ND + threadIdx.x * 4;
    const float4 a = *(const float4*)&x0[i];
    const float4 g = *(const float4*)&r[i];
    float4 o;
    o.x = a.x * (g.x + 1.0f); o.y = a.y * (g.y + 1.0f);
    o.z = a.z * (g.z + 1.0f); o.w = a.w * (g.w + 1.0f);
    *(float4*)&x[i] = o;
}

// ---------------------------------------------------------------------------
// Fused causal scores + softmax: one block per (b,h,q) row
// ---------------------------------------------------------------------------
__global__ __launch_bounds__(256) void scores_softmax_kernel(
    const float* __restrict__ q, const float* __restrict__ k, float* __restrict__ probs)
{
    __shared__ float red[256];
    const int qi = blockIdx.x, h = blockIdx.y, b = blockIdx.z;
    const int t  = threadIdx.x;

    float sc = -INFINITY;
    if (t < NS && t <= qi) {
        const float* qp = q + ((size_t)(b * NS + qi)) * ND + h * NHD;
        const float* kp = k + ((size_t)(b * NS + t )) * ND + h * NHD;
        float dot = 0.0f;
#pragma unroll
        for (int d = 0; d < NHD; d += 4) {
            const float4 a = *(const float4*)&qp[d];
            const float4 c = *(const float4*)&kp[d];
            dot += a.x * c.x + a.y * c.y + a.z * c.z + a.w * c.w;
        }
        sc = dot * 0.125f;   // 1/sqrt(64)
    }
    red[t] = sc;
    __syncthreads();
    for (int off = 128; off > 0; off >>= 1) {
        if (t < off) red[t] = fmaxf(red[t], red[t + off]);
        __syncthreads();
    }
    const float mx = red[0];
    __syncthreads();

    float e = 0.0f;
    if (t < NS && t <= qi) e = __expf(sc - mx);
    red[t] = e;
    __syncthreads();
    for (int off = 128; off > 0; off >>= 1) {
        if (t < off) red[t] += red[t + off];
        __syncthreads();
    }
    const float inv = 1.0f / red[0];
    if (t < NS)
        probs[(((size_t)(b * NH + h)) * NS + qi) * NS + t] = e * inv;
}

// o[b,q,:] = (probs @ v) * r   (per-row, thread handles 4 dims of one head)
__global__ __launch_bounds__(256) void av_rgate_kernel(
    const float* __restrict__ probs, const float* __restrict__ v,
    const float* __restrict__ r, float* __restrict__ o)
{
    const int row = blockIdx.x;           // b*NS + q
    const int b   = row / NS;
    const int qi  = row % NS;
    const int d   = threadIdx.x * 4;
    const int h   = d >> 6;               // all 4 dims same head
    const float* pp = probs + (((size_t)(b * NH + h)) * NS + qi) * NS;

    float ax = 0.0f, ay = 0.0f, az = 0.0f, aw = 0.0f;
    for (int kk = 0; kk <= qi; ++kk) {
        const float p = pp[kk];
        const float4 vv = *(const float4*)&v[((size_t)(b * NS + kk)) * ND + d];
        ax += p * vv.x; ay += p * vv.y; az += p * vv.z; aw += p * vv.w;
    }
    const float4 rr = *(const float4*)&r[(size_t)row * ND + d];
    float4 oo;
    oo.x = ax * rr.x; oo.y = ay * rr.y; oo.z = az * rr.z; oo.w = aw * rr.w;
    *(float4*)&o[(size_t)row * ND + d] = oo;
}

// ---------------------------------------------------------------------------
// y = LayerNorm(x [+ res]) * g + b   (block per row, D=1024, thread = 4 elems)
// ---------------------------------------------------------------------------
__global__ __launch_bounds__(256) void add_ln_kernel(
    const float* __restrict__ x, const float* __restrict__ res,
    const float* __restrict__ g, const float* __restrict__ bp,
    float* __restrict__ out)
{
    __shared__ float red[256];
    const int row = blockIdx.x;
    const int t   = threadIdx.x;
    const size_t base = (size_t)row * ND + t * 4;

    float4 vv = *(const float4*)&x[base];
    if (res) {
        const float4 a = *(const float4*)&res[base];
        vv.x += a.x; vv.y += a.y; vv.z += a.z; vv.w += a.w;
    }
    float s  = vv.x + vv.y + vv.z + vv.w;
    float ss = vv.x * vv.x + vv.y * vv.y + vv.z * vv.z + vv.w * vv.w;

    red[t] = s; __syncthreads();
    for (int off = 128; off > 0; off >>= 1) { if (t < off) red[t] += red[t + off]; __syncthreads(); }
    const float mean = red[0] * (1.0f / ND);
    __syncthreads();
    red[t] = ss; __syncthreads();
    for (int off = 128; off > 0; off >>= 1) { if (t < off) red[t] += red[t + off]; __syncthreads(); }
    const float var  = red[0] * (1.0f / ND) - mean * mean;
    const float rstd = rsqrtf(var + 1e-5f);

    const float4 gg = *(const float4*)&g[t * 4];
    const float4 bb = *(const float4*)&bp[t * 4];
    float4 oo;
    oo.x = (vv.x - mean) * rstd * gg.x + bb.x;
    oo.y = (vv.y - mean) * rstd * gg.y + bb.y;
    oo.z = (vv.z - mean) * rstd * gg.z + bb.z;
    oo.w = (vv.w - mean) * rstd * gg.w + bb.w;
    *(float4*)&out[base] = oo;
}

// ---------------------------------------------------------------------------
// Orchestration
// ---------------------------------------------------------------------------
extern "C" void kernel_launch(void* const* d_in, const int* in_sizes, int n_in,
                              void* d_out, int out_size, void* d_ws, size_t ws_size,
                              hipStream_t stream) {
    const int*   ids     = (const int*)  d_in[0];
    const float* tok_emb = (const float*)d_in[1];
    const float* emb_w   = (const float*)d_in[2];
    const float* emb_b   = (const float*)d_in[3];
    const float* pe      = (const float*)d_in[4];
    const float* attn_w  = (const float*)d_in[5];
    const float* attn_b  = (const float*)d_in[6];
    const float* ln_g    = (const float*)d_in[7];
    const float* ln_b    = (const float*)d_in[8];
    const float* ff1_w   = (const float*)d_in[9];
    const float* ff1_b   = (const float*)d_in[10];
    const float* ff2_w   = (const float*)d_in[11];
    const float* ff2_b   = (const float*)d_in[12];
    const float* fln_g   = (const float*)d_in[13];
    const float* fln_b   = (const float*)d_in[14];
    const float* out_w   = (const float*)d_in[15];
    const float* out_b   = (const float*)d_in[16];
    float* out = (float*)d_out;

    // workspace layout (floats)
    float* ws = (float*)d_ws;
    const size_t MD = (size_t)NM * ND;
    float* x0 = ws;                         // also attn-out / final-ln scratch
    float* x  = x0 + MD;
    float* qb = x  + MD;                    // q, later o-proj / ffn2 scratch
    float* kb = qb + MD;
    float* vb = kb + MD;
    float* rb = vb + MD;
    float* hb = rb + MD;                    // ffn hidden  [NM, NF]
    float* pb = hb + (size_t)NM * NF;       // probs [NB,NH,NS,NS]

    const dim3 blk(256);
    const dim3 gD(ND / 64, NM / 128);

    // embedding + multiplicative r-gate
    embed_kernel<<<NM, blk, 0, stream>>>(ids, tok_emb, pe, x0);
    gemm_bf16_wmma<0><<<gD, blk, 0, stream>>>(x0, emb_w, emb_b, qb, ND, ND);
    rgate_kernel<<<NM, blk, 0, stream>>>(x0, qb, x);

    for (int l = 0; l < NL; ++l) {
        const float* w  = attn_w + (size_t)l * 5 * ND * ND;
        const float* bb = attn_b + (size_t)l * 5 * ND;

        gemm_bf16_wmma<0><<<gD, blk, 0, stream>>>(x,  w + 0 * (size_t)ND * ND, bb + 0 * ND, qb, ND, ND);
        gemm_bf16_wmma<0><<<gD, blk, 0, stream>>>(x,  w + 1 * (size_t)ND * ND, bb + 1 * ND, kb, ND, ND);
        gemm_bf16_wmma<0><<<gD, blk, 0, stream>>>(x,  w + 2 * (size_t)ND * ND, bb + 2 * ND, vb, ND, ND);
        gemm_bf16_wmma<0><<<gD, blk, 0, stream>>>(qb, w + 3 * (size_t)ND * ND, bb + 3 * ND, rb, ND, ND);

        { dim3 gs(NS, NH, NB);
          scores_softmax_kernel<<<gs, blk, 0, stream>>>(qb, kb, pb); }
        av_rgate_kernel<<<NM, blk, 0, stream>>>(pb, vb, rb, x0);

        gemm_bf16_wmma<0><<<gD, blk, 0, stream>>>(x0, w + 4 * (size_t)ND * ND, bb + 4 * ND, qb, ND, ND);
        add_ln_kernel<<<NM, blk, 0, stream>>>(x, qb, ln_g + (size_t)(l * 2 + 0) * ND,
                                              ln_b + (size_t)(l * 2 + 0) * ND, x);

        { dim3 g1(NF / 64, NM / 128);
          gemm_bf16_wmma<1><<<g1, blk, 0, stream>>>(x, ff1_w + (size_t)l * ND * NF,
                                                    ff1_b + (size_t)l * NF, hb, ND, NF); }
        gemm_bf16_wmma<0><<<gD, blk, 0, stream>>>(hb, ff2_w + (size_t)l * NF * ND,
                                                  ff2_b + (size_t)l * ND, qb, NF, ND);
        add_ln_kernel<<<NM, blk, 0, stream>>>(x, qb, ln_g + (size_t)(l * 2 + 1) * ND,
                                              ln_b + (size_t)(l * 2 + 1) * ND, x);
    }

    add_ln_kernel<<<NM, blk, 0, stream>>>(x, nullptr, fln_g, fln_b, x0);
    { dim3 go(NV / 64, NM / 128);
      gemm_bf16_wmma<0><<<go, blk, 0, stream>>>(x0, out_w, out_b, out, ND, NV); }
}
